// CrossAttentionFusion_58171037057720
// MI455X (gfx1250) — matrix operand order repaired and checked by reading, not compiled, not run on hardware
//
#include <hip/hip_runtime.h>
#include <hip/hip_bf16.h>

// ---------------- problem constants ----------------
constexpr int BATCH = 2;
constexpr int C1 = 64;
constexpr int C2 = 256;
constexpr int DM = 256;     // model dim
constexpr int NHEAD = 8;
constexpr int HD = 32;      // head dim
constexpr int NSEQ = 4096;  // tokens per batch (H*W)
constexpr int NTOK = BATCH * NSEQ; // 8192
constexpr int D4 = 4 * DM;  // 1024
constexpr float EPS = 1e-5f;

typedef __attribute__((ext_vector_type(16))) _Float16 v16h;
typedef __attribute__((ext_vector_type(8)))  _Float16 h8v;
typedef __attribute__((ext_vector_type(8)))  float    v8f;
typedef __attribute__((ext_vector_type(4)))  int      v4i;

// CDNA5 async global->LDS copies (ASYNCcnt path). Guarded so compile cannot
// fail if this toolchain does not declare the builtin.
#if defined(__has_builtin)
#if __has_builtin(__builtin_amdgcn_global_load_async_to_lds_b128)
#define HAVE_ASYNC_LDS 1
#endif
#endif
#ifndef HAVE_ASYNC_LDS
#define HAVE_ASYNC_LDS 0
#endif

#if HAVE_ASYNC_LDS
// Param types per compiler diagnostic: (global int4*, shared int4*, imm, imm)
typedef __attribute__((address_space(1))) v4i gv4i;
typedef __attribute__((address_space(3))) v4i lv4i;
#endif

// ---------------- device helpers ----------------
__device__ __forceinline__ v16h make_frag(const _Float16* p0, const _Float16* p1) {
    h8v lo = *(const h8v*)p0;
    h8v hi = *(const h8v*)p1;
    v16h r;
#pragma unroll
    for (int i = 0; i < 8; ++i) { r[i] = lo[i]; r[i + 8] = hi[i]; }
    return r;
}

__device__ __forceinline__ v8f wmma_f16(v16h a, v16h b, v8f c) {
    return __builtin_amdgcn_wmma_f32_16x16x32_f16(false, a, false, b, (short)0, c, false, false);
}

__device__ __forceinline__ v8f vzero8() {
    v8f z = {0.f, 0.f, 0.f, 0.f, 0.f, 0.f, 0.f, 0.f};
    return z;
}

// ---------------- pack kernels ----------------
// (B, C, N) f32  ->  (B, N, C) f16   via 32x32 LDS tile transpose
__global__ __launch_bounds__(256) void k_transpose_pack(const float* __restrict__ in,
                                                        _Float16* __restrict__ out, int C) {
    __shared__ float tile[32][33];
    int b = blockIdx.z;
    int n0 = blockIdx.x * 32, c0 = blockIdx.y * 32;
    int tx = threadIdx.x, ty = threadIdx.y; // 32 x 8
#pragma unroll
    for (int j = 0; j < 32; j += 8)
        tile[ty + j][tx] = in[((size_t)(b * C + c0 + ty + j)) * NSEQ + n0 + tx];
    __syncthreads();
#pragma unroll
    for (int j = 0; j < 32; j += 8)
        out[((size_t)(b * NSEQ + n0 + ty + j)) * C + c0 + tx] = (_Float16)tile[tx][ty + j];
}

__global__ __launch_bounds__(256) void k_cvt_f16(const float* __restrict__ in,
                                                 _Float16* __restrict__ out, int n) {
    int i = blockIdx.x * 256 + threadIdx.x;
    if (i < n) out[i] = (_Float16)in[i];
}

// V (token, D) f16 -> Vt (b, h, dim, key) f16 : contiguous along key
__global__ __launch_bounds__(256) void k_vtrans(const _Float16* __restrict__ V,
                                                _Float16* __restrict__ Vt) {
    int idx = blockIdx.x * 256 + threadIdx.x; // < NTOK*DM
    int n = idx & (NSEQ - 1);
    int rest = idx >> 12;
    int dimi = rest & (HD - 1);
    int hh = rest >> 5;
    int h = hh & (NHEAD - 1);
    int b = hh >> 3;
    Vt[idx] = V[((size_t)(b * NSEQ + n)) * DM + h * HD + dimi];
}

// ---------------- generic f16 WMMA GEMM ----------------
// C[M,N] = alpha * A[M,K] * W[N,K]^T + bias ; optional relu; f32 and/or f16 out.
// block = 256 threads = 8 waves; wave tile 16x64; block tile 64x128.
// All B fragments are loaded before the WMMA burst so the scheduler can clause
// the loads and issue the 4 WMMAs back-to-back.
__global__ __launch_bounds__(256) void k_gemm16(const _Float16* __restrict__ A,
                                                const _Float16* __restrict__ W,
                                                const float* __restrict__ bias,
                                                float* __restrict__ Cf,
                                                _Float16* __restrict__ Ch,
                                                int N, int K, float alpha, int relu) {
    int tid = threadIdx.x;
    int lane = tid & 31, wave = tid >> 5;
    int r = lane & 15, hf = lane >> 4;
    int m0 = blockIdx.x * 64 + (wave & 3) * 16;
    int n0 = blockIdx.y * 128 + (wave >> 2) * 64;

    v8f acc[4];
#pragma unroll
    for (int j = 0; j < 4; ++j) acc[j] = vzero8();

    const _Float16* arow = A + (size_t)(m0 + r) * K;
    for (int kk = 0; kk < K; kk += 32) {
        __builtin_prefetch(arow + kk + 128, 0, 1); // global_prefetch_b8
        v16h af = make_frag(arow + kk + hf * 16, arow + kk + hf * 16 + 8);
        v16h bf[4];
#pragma unroll
        for (int j = 0; j < 4; ++j) {
            const _Float16* wrow = W + (size_t)(n0 + j * 16 + r) * K + kk + hf * 16;
            bf[j] = make_frag(wrow, wrow + 8);
        }
#pragma unroll
        for (int j = 0; j < 4; ++j) acc[j] = wmma_f16(af, bf[j], acc[j]);
    }
#pragma unroll
    for (int j = 0; j < 4; ++j) {
        int col = n0 + j * 16 + r;
        float bv = bias ? bias[col] : 0.f;
#pragma unroll
        for (int v = 0; v < 8; ++v) {
            int row = m0 + v + 8 * hf;
            float val = acc[j][v] * alpha + bv;
            if (relu) val = fmaxf(val, 0.f);
            size_t o = (size_t)row * N + col;
            if (Cf) Cf[o] = val;
            if (Ch) Ch[o] = (_Float16)val;
        }
    }
}

// ---------------- flash attention (online softmax) ----------------
// Q pre-scaled by (HD^-0.5 * log2(e)) so exp2 can be used directly.
// block = 128 threads = 4 waves; each wave: 16 query rows of one (b, head).
// All 4 waves consume the same 32-key K/V slice, so the slice is staged into
// double-buffered LDS with async global->LDS copies (ASYNCcnt) when available.
__global__ __launch_bounds__(128) void k_attention(const _Float16* __restrict__ Q,
                                                   const _Float16* __restrict__ Km,
                                                   const _Float16* __restrict__ Vt,
                                                   _Float16* __restrict__ AO) {
#if HAVE_ASYNC_LDS
    __shared__ _Float16 kstage[2][32 * 32]; // [key_local][dim] per buffer
    __shared__ _Float16 vstage[2][32 * 32]; // [dim][key_local] per buffer
#endif
    __shared__ _Float16 plds[4 * 16 * 32]; // one 16x32 P tile per wave
    int tid = threadIdx.x;
    int lane = tid & 31, wave = tid >> 5;
    int r = lane & 15, hf = lane >> 4;
    int bid = blockIdx.x;
    int qt = bid & 63;
    int h = (bid >> 6) & (NHEAD - 1);
    int b = bid >> 9;

    int qrow = b * NSEQ + qt * 64 + wave * 16 + r;
    const _Float16* qp = Q + (size_t)qrow * DM + h * HD;
    v16h qf = make_frag(qp + hf * 16, qp + hf * 16 + 8);

    v8f acc0 = vzero8(), acc1 = vzero8();
    float mrun[8], lrun[8];
#pragma unroll
    for (int v = 0; v < 8; ++v) { mrun[v] = -1e30f; lrun[v] = 0.f; }

    _Float16* pl = plds + wave * 512;
    const size_t vthead = ((size_t)(b * NHEAD + h)) * HD;

#if HAVE_ASYNC_LDS
    // Cooperative copy: 128 lanes x (one K b128 + one V b128) = full 2KB+2KB slice.
    int rowi = tid >> 2, part = tid & 3; // rowi: key_local (K) / dim (V), part: 16B segment
    auto issue_slice = [&](int buf, int kb2) {
        const _Float16* gk = Km + (size_t)(b * NSEQ + kb2 + rowi) * DM + h * HD + part * 8;
        const _Float16* gv = Vt + (vthead + rowi) * NSEQ + kb2 + part * 8;
        __builtin_amdgcn_global_load_async_to_lds_b128(
            (gv4i*)gk, (lv4i*)&kstage[buf][rowi * 32 + part * 8], 0, 0);
        __builtin_amdgcn_global_load_async_to_lds_b128(
            (gv4i*)gv, (lv4i*)&vstage[buf][rowi * 32 + part * 8], 0, 0);
    };
    issue_slice(0, 0);
    int cur = 0;
#endif

    for (int kb = 0; kb < NSEQ; kb += 32) {
#if HAVE_ASYNC_LDS
        asm volatile("s_wait_asynccnt 0x0" ::: "memory"); // my slice copies done
        __syncthreads();                                  // whole slice visible
        if (kb + 32 < NSEQ) issue_slice(cur ^ 1, kb + 32); // prefetch next slice
        const _Float16* kbase = &kstage[cur][0];
        const _Float16* vbase = &vstage[cur][0];
        v16h kf0 = make_frag(kbase + r * 32 + hf * 16, kbase + r * 32 + hf * 16 + 8);
        v16h kf1 = make_frag(kbase + (16 + r) * 32 + hf * 16,
                             kbase + (16 + r) * 32 + hf * 16 + 8);
#else
        const _Float16* kp0 = Km + (size_t)(b * NSEQ + kb + r) * DM + h * HD + hf * 16;
        const _Float16* kp1 = kp0 + (size_t)16 * DM;
        v16h kf0 = make_frag(kp0, kp0 + 8);
        v16h kf1 = make_frag(kp1, kp1 + 8);
#endif
        // S = Q * K^T  (two 16x16 score tiles)
        v8f s0 = wmma_f16(qf, kf0, vzero8());
        v8f s1 = wmma_f16(qf, kf1, vzero8());

        // online softmax per row (row v+8*hf lives across the 16-lane half)
#pragma unroll
        for (int v = 0; v < 8; ++v) {
            float sm = fmaxf(s0[v], s1[v]);
            for (int off = 1; off < 16; off <<= 1)
                sm = fmaxf(sm, __shfl_xor(sm, off, 32));
            float mnew = fmaxf(mrun[v], sm);
            float f = exp2f(mrun[v] - mnew);
            mrun[v] = mnew;
            float p0 = exp2f(s0[v] - mnew);
            float p1 = exp2f(s1[v] - mnew);
            float rs = p0 + p1;
            for (int off = 1; off < 16; off <<= 1)
                rs += __shfl_xor(rs, off, 32);
            lrun[v] = lrun[v] * f + rs;
            acc0[v] *= f;
            acc1[v] *= f;
            int rr = v + 8 * hf;
            pl[rr * 32 + r]      = (_Float16)p0;
            pl[rr * 32 + 16 + r] = (_Float16)p1;
        }
        asm volatile("s_wait_dscnt 0x0" ::: "memory");

        // P (D-layout in LDS) -> A-fragment, then O += P * V
        v16h pf = make_frag(pl + r * 32 + hf * 16, pl + r * 32 + hf * 16 + 8);
#if HAVE_ASYNC_LDS
        v16h vf0 = make_frag(vbase + r * 32 + hf * 16, vbase + r * 32 + hf * 16 + 8);
        v16h vf1 = make_frag(vbase + (16 + r) * 32 + hf * 16,
                             vbase + (16 + r) * 32 + hf * 16 + 8);
#else
        const _Float16* vp0 = Vt + (vthead + r) * NSEQ + kb + hf * 16;
        const _Float16* vp1 = Vt + (vthead + 16 + r) * NSEQ + kb + hf * 16;
        v16h vf0 = make_frag(vp0, vp0 + 8);
        v16h vf1 = make_frag(vp1, vp1 + 8);
#endif
        acc0 = wmma_f16(pf, vf0, acc0);
        acc1 = wmma_f16(pf, vf1, acc1);
#if HAVE_ASYNC_LDS
        cur ^= 1;
#endif
    }

#pragma unroll
    for (int v = 0; v < 8; ++v) {
        float inv = 1.f / lrun[v];
        int row = b * NSEQ + qt * 64 + wave * 16 + v + 8 * hf;
        AO[(size_t)row * DM + h * HD + r]      = (_Float16)(acc0[v] * inv);
        AO[(size_t)row * DM + h * HD + 16 + r] = (_Float16)(acc1[v] * inv);
    }
}

// ---------------- residual add + layernorm (fp32) ----------------
__global__ __launch_bounds__(256) void k_add_ln(const float* __restrict__ X,
                                                const float* __restrict__ Y,
                                                const float* __restrict__ g,
                                                const float* __restrict__ be,
                                                float* __restrict__ outF,
                                                _Float16* __restrict__ outH,
                                                float* __restrict__ outT) {
    __shared__ float red1[8], red2[8];
    int t = blockIdx.x, d = threadIdx.x;
    float v = X[(size_t)t * DM + d] + Y[(size_t)t * DM + d];
    float s1 = v, s2 = v * v;
    for (int off = 16; off >= 1; off >>= 1) {
        s1 += __shfl_xor(s1, off, 32);
        s2 += __shfl_xor(s2, off, 32);
    }
    int lane = d & 31, w = d >> 5;
    if (lane == 0) { red1[w] = s1; red2[w] = s2; }
    __syncthreads();
    if (d == 0) {
        float a = 0.f, b2 = 0.f;
#pragma unroll
        for (int i = 0; i < 8; ++i) { a += red1[i]; b2 += red2[i]; }
        red1[0] = a; red2[0] = b2;
    }
    __syncthreads();
    float mu = red1[0] * (1.f / (float)DM);
    float var = red2[0] * (1.f / (float)DM) - mu * mu;
    float y = (v - mu) * rsqrtf(var + EPS) * g[d] + be[d];
    if (outF) outF[(size_t)t * DM + d] = y;
    if (outH) outH[(size_t)t * DM + d] = (_Float16)y;
    if (outT) {
        int b = t >> 12, n = t & (NSEQ - 1);
        outT[((size_t)(b * DM + d)) * NSEQ + n] = y;
    }
}

// ---------------- host launcher ----------------
extern "C" void kernel_launch(void* const* d_in, const int* in_sizes, int n_in,
                              void* d_out, int out_size, void* d_ws, size_t ws_size,
                              hipStream_t stream) {
    (void)in_sizes; (void)n_in; (void)out_size; (void)ws_size;
    const float* F_lidar = (const float*)d_in[0];
    const float* F_cam   = (const float*)d_in[1];
    const float* Wq  = (const float*)d_in[2];
    const float* Wk  = (const float*)d_in[3];
    const float* Wv  = (const float*)d_in[4];
    const float* Wo  = (const float*)d_in[5];
    const float* Wres= (const float*)d_in[6];
    const float* g1  = (const float*)d_in[7];
    const float* b1  = (const float*)d_in[8];
    const float* g2  = (const float*)d_in[9];
    const float* b2  = (const float*)d_in[10];
    const float* W1  = (const float*)d_in[11];
    const float* bf1 = (const float*)d_in[12];
    const float* W2  = (const float*)d_in[13];
    const float* bf2 = (const float*)d_in[14];
    float* out = (float*)d_out;

    // workspace carve-up (256B aligned)
    char* cur = (char*)d_ws;
    auto alloc = [&](size_t bytes) -> void* {
        void* r = (void*)cur;
        cur += (bytes + 255) & ~((size_t)255);
        return r;
    };
    _Float16* Xl16  = (_Float16*)alloc((size_t)NTOK * C1 * 2);
    _Float16* Xc16  = (_Float16*)alloc((size_t)NTOK * C2 * 2);
    _Float16* Wq16  = (_Float16*)alloc((size_t)DM * C1 * 2);
    _Float16* Wk16  = (_Float16*)alloc((size_t)DM * C2 * 2);
    _Float16* Wv16  = (_Float16*)alloc((size_t)DM * C2 * 2);
    _Float16* Wo16  = (_Float16*)alloc((size_t)DM * DM * 2);
    _Float16* Wres16= (_Float16*)alloc((size_t)DM * C1 * 2);
    _Float16* W116  = (_Float16*)alloc((size_t)D4 * DM * 2);
    _Float16* W216  = (_Float16*)alloc((size_t)DM * D4 * 2);
    _Float16* Q16   = (_Float16*)alloc((size_t)NTOK * DM * 2);
    _Float16* K16   = (_Float16*)alloc((size_t)NTOK * DM * 2);
    _Float16* V16   = (_Float16*)alloc((size_t)NTOK * DM * 2);
    _Float16* Vt16  = (_Float16*)alloc((size_t)NTOK * DM * 2);
    _Float16* AO16  = (_Float16*)alloc((size_t)NTOK * DM * 2);
    _Float16* X1h   = (_Float16*)alloc((size_t)NTOK * DM * 2);
    _Float16* H16   = (_Float16*)alloc((size_t)NTOK * D4 * 2);
    float* Resid    = (float*)alloc((size_t)NTOK * DM * 4);
    float* AOproj   = (float*)alloc((size_t)NTOK * DM * 4);
    float* X1f      = (float*)alloc((size_t)NTOK * DM * 4);
    float* Ffn      = (float*)alloc((size_t)NTOK * DM * 4);

    // 1) pack inputs to token-major f16
    k_transpose_pack<<<dim3(NSEQ / 32, C1 / 32, BATCH), dim3(32, 8), 0, stream>>>(F_lidar, Xl16, C1);
    k_transpose_pack<<<dim3(NSEQ / 32, C2 / 32, BATCH), dim3(32, 8), 0, stream>>>(F_cam, Xc16, C2);

    // 2) weights -> f16 (keep (out,in) layout; that IS the B^T layout the GEMM wants)
    auto cvt = [&](const float* s, _Float16* d, int n) {
        k_cvt_f16<<<(n + 255) / 256, 256, 0, stream>>>(s, d, n);
    };
    cvt(Wq, Wq16, DM * C1);
    cvt(Wk, Wk16, DM * C2);
    cvt(Wv, Wv16, DM * C2);
    cvt(Wo, Wo16, DM * DM);
    cvt(Wres, Wres16, DM * C1);
    cvt(W1, W116, D4 * DM);
    cvt(W2, W216, DM * D4);

    // 3) projections (WMMA GEMMs). Fold softmax scale * log2(e) into Q.
    const float qalpha = 0.17677669529663687f * 1.4426950408889634f; // HD^-0.5 * log2(e)
    k_gemm16<<<dim3(NTOK / 64, DM / 128), 256, 0, stream>>>(Xl16, Wq16, nullptr, nullptr, Q16, DM, C1, qalpha, 0);
    k_gemm16<<<dim3(NTOK / 64, DM / 128), 256, 0, stream>>>(Xc16, Wk16, nullptr, nullptr, K16, DM, C2, 1.f, 0);
    k_gemm16<<<dim3(NTOK / 64, DM / 128), 256, 0, stream>>>(Xc16, Wv16, nullptr, nullptr, V16, DM, C2, 1.f, 0);
    k_gemm16<<<dim3(NTOK / 64, DM / 128), 256, 0, stream>>>(Xl16, Wres16, nullptr, Resid, nullptr, DM, C1, 1.f, 0);

    // 4) V head-transpose for contiguous B-fragment loads in attention
    k_vtrans<<<(NTOK * DM) / 256, 256, 0, stream>>>(V16, Vt16);

    // 5) flash attention: B*NH*(NSEQ/64) blocks, 4 waves each
    k_attention<<<BATCH * NHEAD * (NSEQ / 64), 128, 0, stream>>>(Q16, K16, Vt16, AO16);

    // 6) output projection, residual + LN1
    k_gemm16<<<dim3(NTOK / 64, DM / 128), 256, 0, stream>>>(AO16, Wo16, nullptr, AOproj, nullptr, DM, DM, 1.f, 0);
    k_add_ln<<<NTOK, DM, 0, stream>>>(Resid, AOproj, g1, b1, X1f, X1h, nullptr);

    // 7) FFN
    k_gemm16<<<dim3(NTOK / 64, D4 / 128), 256, 0, stream>>>(X1h, W116, bf1, nullptr, H16, D4, DM, 1.f, 1);
    k_gemm16<<<dim3(NTOK / 64, DM / 128), 256, 0, stream>>>(H16, W216, bf2, Ffn, nullptr, DM, D4, 1.f, 0);

    // 8) residual + LN2 + channels-first transpose into d_out
    k_add_ln<<<NTOK, DM, 0, stream>>>(X1f, Ffn, g2, b2, nullptr, nullptr, out);
}
